// TimeVaryingHMM_18683107737753
// MI455X (gfx1250) — compile-verified, matrix operand order checked by reading, not run on hardware
//
#include <hip/hip_runtime.h>
#include <hip/hip_bf16.h>
#include <math.h>

// ---------------------------------------------------------------------------
// Time-varying HMM log-likelihood on MI455X (gfx1250, wave32, WMMA, async-LDS).
//
// Forward recursion  alpha_t = le_t + logsumexp_i(alpha_{t-1} + logA_t)  is an
// associative composition of log-domain operators B_t[i,j] = logA_t[i,j]+le_t[j].
// We exploit that:
//   prep : pack W[K,K,Dx] into f16 WMMA B-fragment layout, invvar/econst.
//   K1   : 512 independent segments x 256 steps. Per 16-step subtile: WMMA
//          logits -> LDS (f16), row-lse + emissions -> LDS, then serial
//          composition M <- M (x) B_t held distributed (thread (i,j) = M[i,j]).
//          One 32x32 matrix per segment to workspace (2 MB total).
//   K2   : single wave32 spine: alpha <- alpha (x) M_s over 512 steps only,
//          with GLOBAL_LOAD_ASYNC_TO_LDS_B128 double-buffering + s_wait_asynccnt.
// ---------------------------------------------------------------------------

#define HMM_K    32
#define HMM_D    16
#define SEG_T    256
#define LOG2PI_F 1.8378770664093453f
#define NEG_BIG  (-1.0e30f)

typedef __attribute__((ext_vector_type(16))) _Float16 v16h;
typedef __attribute__((ext_vector_type(8)))  float    v8f;

// ---------------------------------------------------------------------------
// Prep: pack W into WMMA B-fragment layout (K padded 16->32 with zeros):
//   lane L in [0,15]  : column n = ct*16+L,      halves 0..7 = d 0..7
//   lane L in [16,31] : column n = ct*16+(L-16), halves 0..7 = d 8..15
// Also invvar = exp(-clip(log_var)), econst = -0.5*(D*log2pi + sum log_var).
// ---------------------------------------------------------------------------
__global__ void __launch_bounds__(256)
hmm_prep_kernel(const float* __restrict__ W, const float* __restrict__ rlv,
                void* __restrict__ WhV, float* __restrict__ invvar,
                float* __restrict__ econst)
{
    _Float16* Wh = (_Float16*)WhV;
    int tid = threadIdx.x;
    for (int idx = tid; idx < 64 * 32 * 16; idx += 256) {
        int ct   = idx >> 9;
        int rem  = idx & 511;
        int lane = rem >> 4;
        int h    = rem & 15;
        int n    = ct * 16 + (lane & 15);
        float v  = 0.0f;
        if (h < 8) {
            int d = (lane < 16) ? h : (8 + h);
            v = W[(size_t)n * HMM_D + d];
        }
        Wh[(size_t)ct * 512 + (size_t)lane * 16 + h] = (_Float16)v;
    }
    if (tid < HMM_K) {
        float s = 0.0f;
        for (int d = 0; d < HMM_D; ++d) {
            float lv = rlv[tid * HMM_D + d];
            lv = fminf(6.0f, fmaxf(-6.0f, lv));
            s += lv;
            invvar[tid * HMM_D + d] = __expf(-lv);
        }
        econst[tid] = -0.5f * ((float)HMM_D * LOG2PI_F + s);
    }
}

// ---------------------------------------------------------------------------
// K1: per-segment operator composition. Block = 1024 threads (32 waves).
// thread (i = tid>>5, j = tid&31) owns M[i,j]; wave i == row i.
// ---------------------------------------------------------------------------
__global__ void __launch_bounds__(1024)
hmm_segment_kernel(const float* __restrict__ X, const float* __restrict__ Y,
                   const float* __restrict__ bmat, const void* __restrict__ WhV,
                   const float* __restrict__ mu, const float* __restrict__ invvar,
                   const float* __restrict__ econst, float* __restrict__ Mseg)
{
    __shared__ _Float16 slog[16 * 1024];   // 32 KB raw logits (f16) for 16 t
    __shared__ float    slse[16 * HMM_K];  // row softmax normalizers
    __shared__ float    sle [16 * HMM_K];  // emissions log_em[t][j]
    __shared__ float    sM  [HMM_K * 33];  // running matrix, padded stride
    __shared__ float    smu [HMM_K * HMM_D];
    __shared__ float    siv [HMM_K * HMM_D];
    __shared__ float    secn[HMM_K];
    __shared__ float    sY  [16 * HMM_D];  // Y rows for this subtile

    const v16h* Bp = (const v16h*)WhV;
    int tid  = threadIdx.x;
    int lane = tid & 31;
    int wave = tid >> 5;
    int i    = wave;          // matrix row owned by this thread
    int j    = lane;          // matrix col owned by this thread
    int seg  = blockIdx.x;

    if (tid < HMM_K * HMM_D) { smu[tid] = mu[tid]; siv[tid] = invvar[tid]; }
    if (tid < HMM_K)         { secn[tid] = econst[tid]; }
    sM[i * 33 + j] = (i == j) ? 0.0f : NEG_BIG;   // log-domain identity
    __syncthreads();

    int rowM = (lane >> 4) * 8;    // C frag: VGPR r -> row r + rowM
    int ncl  = lane & 15;          // C frag: column within tile

    for (int st = 0; st < 16; ++st) {
        int trow = seg * SEG_T + st * 16;

        // ---- stage Y rows for emissions ----
        if (tid < 16 * HMM_D) sY[tid] = Y[(size_t)trow * HMM_D + tid];

        // ---- WMMA logits: 32 waves x 2 column tiles each ----
        int m     = lane & 15;
        int kbase = (lane < 16) ? 0 : 8;
        const float* xr = X + (size_t)(trow + m) * HMM_D + kbase;
        v16h a;
#pragma unroll
        for (int k = 0; k < 8; ++k) a[k] = (_Float16)xr[k];
#pragma unroll
        for (int k = 8; k < 16; ++k) a[k] = (_Float16)0.0f;

#pragma unroll
        for (int q = 0; q < 2; ++q) {
            int ct = wave * 2 + q;             // column tile 0..63
            v16h bf = Bp[(size_t)ct * 32 + lane];
            v8f c = {};
            c = __builtin_amdgcn_wmma_f32_16x16x32_f16(
                    false, a, false, bf, (short)0, c, false, false);
            int n = ct * 16 + ncl;             // global column = ii*32 + jj
            float bias = bmat[n];
#pragma unroll
            for (int r = 0; r < 8; ++r)
                slog[(rowM + r) * 1024 + n] = (_Float16)(c[r] + bias);
        }
        __syncthreads();

        // ---- per-(row, state) lse over 32 j's, and emission le[row][state] ----
        if (tid < 512) {
            int row = tid >> 5;
            int k   = tid & 31;
            const _Float16* p = &slog[row * 1024 + k * 32];
            float mx = (float)p[0];
#pragma unroll
            for (int jj = 1; jj < 32; ++jj) mx = fmaxf(mx, (float)p[jj]);
            float s = 0.0f;
#pragma unroll
            for (int jj = 0; jj < 32; ++jj) s += __expf((float)p[jj] - mx);
            slse[row * HMM_K + k] = mx + __logf(s);

            float quad = 0.0f;
#pragma unroll
            for (int d = 0; d < HMM_D; ++d) {
                float dv = sY[row * HMM_D + d] - smu[k * HMM_D + d];
                quad += dv * dv * siv[k * HMM_D + d];
            }
            sle[row * HMM_K + k] = secn[k] - 0.5f * quad;
        }
        __syncthreads();

        // ---- serial composition over the 16 steps of this subtile ----
        int tstart = (seg == 0 && st == 0) ? 1 : 0;   // t=0 belongs to alpha0
        for (int tl = tstart; tl < 16; ++tl) {
            const _Float16* lp = &slog[tl * 1024 + j];  // [k*32] strided
            const float* lsp = &slse[tl * HMM_K];
            const float* mrp = &sM[i * 33];
            float x[32];
#pragma unroll
            for (int k = 0; k < 32; ++k)
                x[k] = mrp[k] - lsp[k] + (float)lp[k * 32];
            // tree max
            float t16[16], t8[8], t4[4];
#pragma unroll
            for (int k = 0; k < 16; ++k) t16[k] = fmaxf(x[2*k], x[2*k+1]);
#pragma unroll
            for (int k = 0; k < 8;  ++k) t8[k]  = fmaxf(t16[2*k], t16[2*k+1]);
#pragma unroll
            for (int k = 0; k < 4;  ++k) t4[k]  = fmaxf(t8[2*k], t8[2*k+1]);
            float mx = fmaxf(fmaxf(t4[0], t4[1]), fmaxf(t4[2], t4[3]));
            // tree sum of exps
            float e16[16], e8[8], e4[4];
#pragma unroll
            for (int k = 0; k < 16; ++k)
                e16[k] = __expf(x[2*k] - mx) + __expf(x[2*k+1] - mx);
#pragma unroll
            for (int k = 0; k < 8;  ++k) e8[k] = e16[2*k] + e16[2*k+1];
#pragma unroll
            for (int k = 0; k < 4;  ++k) e4[k] = e8[2*k] + e8[2*k+1];
            float s = (e4[0] + e4[1]) + (e4[2] + e4[3]);

            float nm = sle[tl * HMM_K + j] + mx + __logf(s);
            __syncthreads();
            sM[i * 33 + j] = nm;
            __syncthreads();
        }
    }

    Mseg[(size_t)seg * 1024 + i * 32 + j] = sM[i * 33 + j];
}

// ---------------------------------------------------------------------------
// K2: spine scan over segment matrices. One wave32, lane j = state j.
// Segment matrices double-buffered into LDS with async loads (8 x b128 / step).
// ---------------------------------------------------------------------------
__global__ void __launch_bounds__(32)
hmm_spine_kernel(const float* __restrict__ Y, const float* __restrict__ init_logits,
                 const float* __restrict__ mu, const float* __restrict__ invvar,
                 const float* __restrict__ econst, const float* __restrict__ Mseg,
                 float* __restrict__ out, int nseg)
{
    __shared__ float sbuf[2][1024];   // double-buffered 32x32 segment matrix

    int j = threadIdx.x;

    // alpha0 = log_softmax(init_logits) + log_em[0]
    float il = init_logits[j];
    float mx0 = il;
#pragma unroll
    for (int off = 16; off; off >>= 1) mx0 = fmaxf(mx0, __shfl_xor(mx0, off, 32));
    float s0 = __expf(il - mx0);
#pragma unroll
    for (int off = 16; off; off >>= 1) s0 += __shfl_xor(s0, off, 32);
    float quad = 0.0f;
#pragma unroll
    for (int d = 0; d < HMM_D; ++d) {
        float dv = Y[d] - mu[j * HMM_D + d];
        quad += dv * dv * invvar[j * HMM_D + d];
    }
    float alpha = (il - mx0 - __logf(s0)) + (econst[j] - 0.5f * quad);

    // issue async load of segment 0 into buffer 0:
    // lane j covers bytes j*16 + k*512 of the 4 KB matrix.
    {
        unsigned lbase = (unsigned)(size_t)(&sbuf[0][0]);
        unsigned long long g0 = (unsigned long long)(Mseg) + (unsigned)(j * 16);
#pragma unroll
        for (int k = 0; k < 8; ++k) {
            unsigned la = lbase + (unsigned)(j * 16 + k * 512);
            unsigned long long ga = g0 + (unsigned)(k * 512);
            asm volatile("global_load_async_to_lds_b128 %0, %1, off"
                         :: "v"(la), "v"(ga) : "memory");
        }
    }

    for (int s = 0; s < nseg; ++s) {
        if (s + 1 < nseg) {
            int nb = (s + 1) & 1;
            unsigned lbase = (unsigned)(size_t)(&sbuf[nb][0]);
            unsigned long long g0 = (unsigned long long)(Mseg)
                                  + (size_t)(s + 1) * 4096 + (unsigned)(j * 16);
#pragma unroll
            for (int k = 0; k < 8; ++k) {
                unsigned la = lbase + (unsigned)(j * 16 + k * 512);
                unsigned long long ga = g0 + (unsigned)(k * 512);
                asm volatile("global_load_async_to_lds_b128 %0, %1, off"
                             :: "v"(la), "v"(ga) : "memory");
            }
            // 8 new in flight; wait until only those remain -> seg s is in LDS
            asm volatile("s_wait_asynccnt 8" ::: "memory");
        } else {
            asm volatile("s_wait_asynccnt 0" ::: "memory");
        }

        const float* mrow = &sbuf[s & 1][0];
        float x[32];
#pragma unroll
        for (int k2 = 0; k2 < 32; ++k2)
            x[k2] = mrow[k2 * 32 + j] + __shfl(alpha, k2, 32);
        float t16[16], t8[8], t4[4];
#pragma unroll
        for (int k2 = 0; k2 < 16; ++k2) t16[k2] = fmaxf(x[2*k2], x[2*k2+1]);
#pragma unroll
        for (int k2 = 0; k2 < 8;  ++k2) t8[k2]  = fmaxf(t16[2*k2], t16[2*k2+1]);
#pragma unroll
        for (int k2 = 0; k2 < 4;  ++k2) t4[k2]  = fmaxf(t8[2*k2], t8[2*k2+1]);
        float mx = fmaxf(fmaxf(t4[0], t4[1]), fmaxf(t4[2], t4[3]));
        float e16[16], e8[8], e4[4];
#pragma unroll
        for (int k2 = 0; k2 < 16; ++k2)
            e16[k2] = __expf(x[2*k2] - mx) + __expf(x[2*k2+1] - mx);
#pragma unroll
        for (int k2 = 0; k2 < 8;  ++k2) e8[k2] = e16[2*k2] + e16[2*k2+1];
#pragma unroll
        for (int k2 = 0; k2 < 4;  ++k2) e4[k2] = e8[2*k2] + e8[2*k2+1];
        float ssum = (e4[0] + e4[1]) + (e4[2] + e4[3]);
        alpha = mx + __logf(ssum);
    }

    // final logsumexp over states
    float mx = alpha;
#pragma unroll
    for (int off = 16; off; off >>= 1) mx = fmaxf(mx, __shfl_xor(mx, off, 32));
    float s = __expf(alpha - mx);
#pragma unroll
    for (int off = 16; off; off >>= 1) s += __shfl_xor(s, off, 32);
    if (j == 0) out[0] = mx + __logf(s);
}

// ---------------------------------------------------------------------------
// Host. Workspace: Wh (64 KB) | invvar (2 KB) | econst | Mseg (nseg*4 KB ~ 2 MB)
// ---------------------------------------------------------------------------
extern "C" void kernel_launch(void* const* d_in, const int* in_sizes, int n_in,
                              void* d_out, int out_size, void* d_ws, size_t ws_size,
                              hipStream_t stream)
{
    const float* X    = (const float*)d_in[0];
    const float* Y    = (const float*)d_in[1];
    const float* init = (const float*)d_in[2];
    const float* W    = (const float*)d_in[3];
    const float* bmat = (const float*)d_in[4];
    const float* mu   = (const float*)d_in[5];
    const float* rlv  = (const float*)d_in[6];
    float* out = (float*)d_out;

    int T = in_sizes[0] / HMM_D;
    int nseg = (T + SEG_T - 1) / SEG_T;

    char* ws = (char*)d_ws;
    size_t off = 0;
    void*  Wh     = (void*) (ws + off); off += (size_t)64 * 32 * 16 * 2;  // 64 KB
    float* invvar = (float*)(ws + off); off += (size_t)HMM_K * HMM_D * 4;
    float* econst = (float*)(ws + off); off += 256;
    float* Mseg   = (float*)(ws + off); off += (size_t)nseg * 1024 * 4;

    hmm_prep_kernel<<<1, 256, 0, stream>>>(W, rlv, Wh, invvar, econst);
    hmm_segment_kernel<<<nseg, 1024, 0, stream>>>(X, Y, bmat, Wh, mu,
                                                  invvar, econst, Mseg);
    hmm_spine_kernel<<<1, 32, 0, stream>>>(Y, init, mu, invvar, econst,
                                           Mseg, out, nseg);
}
